// ConvAttnLSTMCell_35416300323420
// MI455X (gfx1250) — compile-verified, matrix operand order checked by reading, not run on hardware
//
#include <hip/hip_runtime.h>
#include <cmath>

typedef __attribute__((ext_vector_type(16))) _Float16 v16h;
typedef __attribute__((ext_vector_type(8)))  _Float16 v8h;
typedef __attribute__((ext_vector_type(8)))  float    v8f;
typedef __attribute__((ext_vector_type(4)))  float    v4f;

#define NB    512           // batch
#define HW    64            // H*W
#define NH_   8
#define THD_  512
#define MEM_  8

// ---------------------------------------------------------------------------
// gfx1250 hardware transcendentals
// ---------------------------------------------------------------------------
__device__ __forceinline__ float fast_tanh(float x) {
#if __has_builtin(__builtin_amdgcn_tanhf)
  return __builtin_amdgcn_tanhf(x);
#elif __has_builtin(__builtin_amdgcn_tanh_f32)
  return __builtin_amdgcn_tanh_f32(x);
#else
  float y;
  // v_tanh_f32 is a TRANS32 op: keep a v_nop after it per the hazard rules
  asm volatile("v_tanh_f32 %0, %1\n\tv_nop" : "=v"(y) : "v"(x));
  return y;
#endif
}
__device__ __forceinline__ float fast_rcp(float x) {
#if __has_builtin(__builtin_amdgcn_rcpf)
  return __builtin_amdgcn_rcpf(x);
#else
  return 1.f / x;
#endif
}

// ---------------------------------------------------------------------------
// fp32 -> f16 weight repack (weights then live in L2, shared by all 512 WGs)
// ---------------------------------------------------------------------------
__global__ __launch_bounds__(256) void repack_f16(const float* __restrict__ src,
                                                  _Float16* __restrict__ dst, int n) {
  int i = blockIdx.x * 256 + threadIdx.x;
  if (i < n) dst[i] = (_Float16)src[i];
}

// ---------------------------------------------------------------------------
// 3x3 SAME conv as implicit GEMM with v_wmma_f32_16x16x32_f16.
// One workgroup (8 wave32) per batch image; full f16 im2col tile in LDS
// (CDNA5 320KB WGP LDS). 2x2 quad of 16x16 D-tiles per wave:
// 4 WMMAs per (2 A + 2 B) fragment loads. act_gates=1 fuses LSTM gate
// nonlinearities via hardware v_tanh (sigmoid = 0.5*tanh(x/2)+0.5).
// Layouts: inA/inB [B][cin][64] f32, W [cout][cin*9] f16, out [B][cout][64] f32
// ---------------------------------------------------------------------------
__global__ __launch_bounds__(256) void conv3x3_wmma(
    const float* __restrict__ inA, const float* __restrict__ inB,
    int cinA, int cinB,
    const _Float16* __restrict__ W, const float* __restrict__ bias,
    float* __restrict__ outp, int cout, int act_gates)
{
  const int b    = blockIdx.x;
  const int tid  = threadIdx.x;
  const int cin  = cinA + cinB;
  const int K    = cin * 9;
  const int Kpad = K + 8;                       // pad -> conflict-free ds_load_b128

  extern __shared__ char smem[];
  float*    raw = (float*)smem;                 // [cin][64]
  _Float16* im  = (_Float16*)(smem + (size_t)cin * HW * 4); // [64][Kpad]

  // stage raw activations (coalesced)
  for (int t = tid; t < cin * HW; t += 256) {
    int ci = t >> 6, sp = t & 63;
    float v;
    if (ci < cinA) v = inA[((size_t)b * cinA + ci) * HW + sp];
    else           v = inB[((size_t)b * cinB + (ci - cinA)) * HW + sp];
    raw[t] = v;
  }
  __syncthreads();

  // build f16 im2col: row m = spatial, col k = ci*9 + tap (matches [co][ci][3][3])
  {
    int m = tid / K, k = tid - m * K;           // one div; then incremental
    while (m < HW) {
      int ci = k / 9, tap = k - ci * 9;         // div by constant -> mul/shift
      int dy = tap / 3 - 1, dx = tap - (tap / 3) * 3 - 1;
      int yy = (m >> 3) + dy, xx = (m & 7) + dx;
      float v = 0.f;
      if ((unsigned)yy < 8u && (unsigned)xx < 8u) v = raw[ci * HW + yy * 8 + xx];
      im[m * Kpad + k] = (_Float16)v;
      k += 256;
      if (k >= K) { k -= K; ++m; }
    }
  }
  __syncthreads();

  const int wv    = tid >> 5;
  const int lane  = tid & 31;
  const int sel   = lane >> 4;                  // lane group (ISA 7.12.2 frag layout)
  const int nfrac = lane & 15;
  const int nquads = cout >> 4;                 // (2 M-pairs) * (cout/32 N-pairs)

  for (int p = wv; p < nquads; p += 8) {
    const int mq = p & 1;
    const int nq = p >> 1;
    const int m0 = mq * 32 + nfrac;             // A rows m0, m0+16 (lane%16 = M)
    const int n0 = nq * 32 + nfrac;             // B cols n0, n0+16 (lane%16 = N)
    const _Float16* arow0 = im + (size_t)m0 * Kpad;
    const _Float16* arow1 = arow0 + (size_t)16 * Kpad;
    const _Float16* w0    = W + (size_t)n0 * K;
    const _Float16* w1    = w0 + (size_t)16 * K;

    v8f acc00 = {}, acc01 = {}, acc10 = {}, acc11 = {};
    for (int k0 = 0; k0 < K; k0 += 32) {
      __builtin_prefetch(w0 + k0 + 64, 0, 1);   // global_prefetch on weight stream
      __builtin_prefetch(w1 + k0 + 64, 0, 1);
      v8h a0lo = *(const v8h*)(arow0 + k0 + sel * 8);
      v8h a0hi = *(const v8h*)(arow0 + k0 + 16 + sel * 8);
      v8h a1lo = *(const v8h*)(arow1 + k0 + sel * 8);
      v8h a1hi = *(const v8h*)(arow1 + k0 + 16 + sel * 8);
      v8h b0lo = *(const v8h*)(w0 + k0 + sel * 8);
      v8h b0hi = *(const v8h*)(w0 + k0 + 16 + sel * 8);
      v8h b1lo = *(const v8h*)(w1 + k0 + sel * 8);
      v8h b1hi = *(const v8h*)(w1 + k0 + 16 + sel * 8);
      v16h a0, a1, bb0, bb1;
#pragma unroll
      for (int i = 0; i < 8; ++i) {
        a0[i] = a0lo[i];  a0[8 + i] = a0hi[i];
        a1[i] = a1lo[i];  a1[8 + i] = a1hi[i];
        bb0[i] = b0lo[i]; bb0[8 + i] = b0hi[i];
        bb1[i] = b1lo[i]; bb1[8 + i] = b1hi[i];
      }
      acc00 = __builtin_amdgcn_wmma_f32_16x16x32_f16(false, a0, false, bb0, (short)0, acc00, false, false);
      acc01 = __builtin_amdgcn_wmma_f32_16x16x32_f16(false, a0, false, bb1, (short)0, acc01, false, false);
      acc10 = __builtin_amdgcn_wmma_f32_16x16x32_f16(false, a1, false, bb0, (short)0, acc10, false, false);
      acc11 = __builtin_amdgcn_wmma_f32_16x16x32_f16(false, a1, false, bb1, (short)0, acc11, false, false);
    }

    // epilogue: bias + optional fused gate activation (branchless tanh form),
    // contiguous 32B stores straight from the C-layout (M = sel*8 + vgpr idx)
    const int   nb0 = n0, nb1 = n0 + 16;
    const float bv0 = bias[nb0], bv1 = bias[nb1];
    // gate id per 64-channel block: i,f,o = sigmoid; g = tanh; a = sigmoid
    const int  ga = (nq * 32) >> 6, gb = (nq * 32 + 16) >> 6;
    const float sm0 = (ga == 3) ? 1.f : 0.5f, am0 = sm0, bm0 = (ga == 3) ? 0.f : 0.5f;
    const float sm1 = (gb == 3) ? 1.f : 0.5f, am1 = sm1, bm1 = (gb == 3) ? 0.f : 0.5f;

    v8f cols[4] = {acc00, acc01, acc10, acc11};
    const int   coln[4]  = {nb0, nb1, nb0, nb1};
    const float colb[4]  = {bv0, bv1, bv0, bv1};
    const float colsm[4] = {sm0, sm1, sm0, sm1};
    const float colam[4] = {am0, am1, am0, am1};
    const float colbm[4] = {bm0, bm1, bm0, bm1};
    const int   rowm[4]  = {mq * 32, mq * 32, mq * 32 + 16, mq * 32 + 16};
#pragma unroll
    for (int c = 0; c < 4; ++c) {
      float r[8];
#pragma unroll
      for (int i = 0; i < 8; ++i) {
        float x = cols[c][i] + colb[c];
        if (act_gates) x = colam[c] * fast_tanh(colsm[c] * x) + colbm[c];
        r[i] = x;
      }
      float* o = outp + ((size_t)b * cout + coln[c]) * HW + rowm[c] + sel * 8;
      v4f t0 = {r[0], r[1], r[2], r[3]};
      v4f t1 = {r[4], r[5], r[6], r[7]};
      *(v4f*)o = t0;
      *((v4f*)o + 1) = t1;
    }
  }
}

// ---------------------------------------------------------------------------
// Attention: one wave32 per (b, head). Streaming, single pass over concat_k/v
// fused with pos_w add + k_ret/v_ret output writes. Softmax over MEM=8.
// ---------------------------------------------------------------------------
__global__ __launch_bounds__(256) void attn_kernel(
    const float* __restrict__ kqv,        // [B][192][64]  (k|q|v channel blocks)
    const float* __restrict__ ck,         // [B][8][8][512]
    const float* __restrict__ cv,
    const unsigned char* __restrict__ amask, // [B*NH][1][8] bool
    const float* __restrict__ pos_w,      // [8][4096]
    const float* __restrict__ pos_b,      // [8][8]
    float* __restrict__ kret,             // [B][8][8][512]
    float* __restrict__ vret,
    float* __restrict__ attn_out)         // [B][64][64]
{
  const int b = blockIdx.x;
  const int h = threadIdx.x >> 5;
  const int lane = threadIdx.x & 31;
  const float scale = 0.04419417382415922f; // 1/sqrt(512)
  const float* kqvb = kqv + (size_t)b * 192 * HW;
  const float* pwh = pos_w + h * THD_;

  float q[16];
#pragma unroll
  for (int j = 0; j < 16; ++j) {
    int d = lane + 32 * j;
    q[j] = kqvb[(64 + h * 8 + (d >> 6)) * HW + (d & 63)];
  }

  float sc[8];
#pragma unroll
  for (int s = 0; s < 8; ++s) {
    float part = 0.f;
#pragma unroll
    for (int j = 0; j < 16; ++j) {
      int d = lane + 32 * j;
      float kv;
      if (s < 7) kv = ck[(((size_t)b * MEM_ + (s + 1)) * NH_ + h) * THD_ + d];
      else       kv = kqvb[(h * 8 + (d >> 6)) * HW + (d & 63)];
      kv += pwh[s * 4096 + d];                     // k_full = k + pos_w
      kret[(((size_t)b * MEM_ + s) * NH_ + h) * THD_ + d] = kv;
      part += q[j] * kv;
    }
#pragma unroll
    for (int off = 16; off > 0; off >>= 1) part += __shfl_down(part, off, 32);
    part = __shfl(part, 0, 32);
    float msk = (s == 7) ? 5.0f
              : (amask[(b * NH_ + h) * MEM_ + s] ? -INFINITY : 0.f);
    sc[s] = part * scale + pos_b[s * NH_ + h] + msk;
  }

  float mx = sc[0];
#pragma unroll
  for (int s = 1; s < 8; ++s) mx = fmaxf(mx, sc[s]);
  float wgt[8], sum = 0.f;
#pragma unroll
  for (int s = 0; s < 8; ++s) { wgt[s] = __expf(sc[s] - mx); sum += wgt[s]; }
  float inv = fast_rcp(sum);
#pragma unroll
  for (int s = 0; s < 8; ++s) wgt[s] *= inv;

  float acc[16];
#pragma unroll
  for (int j = 0; j < 16; ++j) acc[j] = 0.f;
#pragma unroll
  for (int s = 0; s < 8; ++s) {
#pragma unroll
    for (int j = 0; j < 16; ++j) {
      int d = lane + 32 * j;
      float vv;
      if (s < 7) vv = cv[(((size_t)b * MEM_ + (s + 1)) * NH_ + h) * THD_ + d];
      else       vv = kqvb[(128 + h * 8 + (d >> 6)) * HW + (d & 63)];
      vret[(((size_t)b * MEM_ + s) * NH_ + h) * THD_ + d] = vv;
      acc[j] += wgt[s] * vv;
    }
  }
#pragma unroll
  for (int j = 0; j < 16; ++j) {
    int d = lane + 32 * j;
    attn_out[(size_t)b * 4096 + (h * 8 + (d >> 6)) * HW + (d & 63)] = acc[j];
  }
}

// ---------------------------------------------------------------------------
// Residual + LayerNorm(E,H,W) + LSTM state update. One block per batch.
// ---------------------------------------------------------------------------
__global__ __launch_bounds__(256) void finalize_kernel(
    const float* __restrict__ outpre, const float* __restrict__ input,
    const float* __restrict__ gates,   // [B][320][64] activated i,f,o,g,a
    const float* __restrict__ c_cur,
    const float* __restrict__ ln_w, const float* __restrict__ ln_b,
    float* __restrict__ h_next, float* __restrict__ c_next)
{
  __shared__ float r1[256], r2[256];
  const int b = blockIdx.x, tid = threadIdx.x;
  const size_t base = (size_t)b * 4096;
  float v[16], s = 0.f;
#pragma unroll
  for (int j = 0; j < 16; ++j) {
    int idx = tid + 256 * j;
    v[j] = outpre[base + idx] + input[base + idx];   // residual (C==E)
    s += v[j];
  }
  r1[tid] = s; __syncthreads();
  for (int o = 128; o > 0; o >>= 1) { if (tid < o) r1[tid] += r1[tid + o]; __syncthreads(); }
  const float mu = r1[0] * (1.f / 4096.f);
  float s2 = 0.f;
#pragma unroll
  for (int j = 0; j < 16; ++j) { float d = v[j] - mu; s2 += d * d; }
  r2[tid] = s2; __syncthreads();
  for (int o = 128; o > 0; o >>= 1) { if (tid < o) r2[tid] += r2[tid + o]; __syncthreads(); }
  const float invstd = rsqrtf(r2[0] * (1.f / 4096.f) + 1e-5f);

  const float* gb = gates + (size_t)b * 20480;
#pragma unroll
  for (int j = 0; j < 16; ++j) {
    int idx = tid + 256 * j;
    float o = (v[j] - mu) * invstd * ln_w[idx] + ln_b[idx];
    float gi = gb[idx], gf = gb[4096 + idx], go = gb[8192 + idx];
    float gg = gb[12288 + idx], ga = gb[16384 + idx];
    float cn = gf * c_cur[base + idx] + gi * gg + ga * fast_tanh(o);
    c_next[base + idx] = cn;
    h_next[base + idx] = go * fast_tanh(cn);
  }
}

// ---------------------------------------------------------------------------
extern "C" void kernel_launch(void* const* d_in, const int* in_sizes, int n_in,
                              void* d_out, int out_size, void* d_ws, size_t ws_size,
                              hipStream_t stream) {
  const float* input  = (const float*)d_in[0];
  const float* h_cur  = (const float*)d_in[1];
  const float* c_cur  = (const float*)d_in[2];
  const float* ck     = (const float*)d_in[3];
  const float* cv     = (const float*)d_in[4];
  const unsigned char* amask = (const unsigned char*)d_in[5]; // bool mask
  const float* conv_w = (const float*)d_in[6];
  const float* conv_b = (const float*)d_in[7];
  const float* proj_w = (const float*)d_in[8];
  const float* proj_b = (const float*)d_in[9];
  const float* out_w  = (const float*)d_in[10];
  const float* out_b  = (const float*)d_in[11];
  const float* ln_w   = (const float*)d_in[12];
  const float* ln_b   = (const float*)d_in[13];
  const float* pos_w  = (const float*)d_in[14];
  const float* pos_b  = (const float*)d_in[15];

  // workspace layout (bytes, 256-aligned)
  char* ws = (char*)d_ws;
  _Float16* wgate = (_Float16*)(ws + 0);            // 320*1152 f16
  _Float16* wproj = (_Float16*)(ws + 737280);       // 192*576  f16
  _Float16* wout  = (_Float16*)(ws + 958464);       // 64*576   f16
  float* gates    = (float*)(ws + 1032192);         // [512][320][64]
  float* kqv      = (float*)(ws + 42975232);        // [512][192][64]
  float* attn     = (float*)(ws + 68141056);        // [512][64][64]
  float* outpre   = (float*)(ws + 76529664);        // [512][64][64]
  if (ws_size < 84918272) return;

  float* outf   = (float*)d_out;
  float* h_next = outf;
  float* c_next = outf + 2097152;
  float* kret   = outf + 4194304;
  float* vret   = outf + 20971520;

  // allow >64KB dynamic LDS (CDNA5 WGP supports up to 320KB per workgroup)
  (void)hipFuncSetAttribute((const void*)conv3x3_wmma,
                            hipFuncAttributeMaxDynamicSharedMemorySize, 181248);

  // 1) weight repack to f16
  repack_f16<<<(368640 + 255) / 256, 256, 0, stream>>>(conv_w, wgate, 368640);
  repack_f16<<<(110592 + 255) / 256, 256, 0, stream>>>(proj_w, wproj, 110592);
  repack_f16<<<(36864  + 255) / 256, 256, 0, stream>>>(out_w,  wout,  36864);

  // 2) gates conv (concat[input,h_cur], K=1152, N=320) + fused gate activations
  //    LDS: 128*64*4 + 64*(1152+8)*2 = 181248 B
  conv3x3_wmma<<<NB, 256, 181248, stream>>>(input, h_cur, 64, 64,
                                            wgate, conv_b, gates, 320, 1);
  // 3) kqv projection conv (K=576, N=192); LDS: 64*64*4 + 64*584*2 = 91136 B
  conv3x3_wmma<<<NB, 256, 91136, stream>>>(input, input, 64, 0,
                                           wproj, proj_b, kqv, 192, 0);
  // 4) attention (streams concat_k/v once, emits k_ret/v_ret + attn map)
  attn_kernel<<<NB, 256, 0, stream>>>(kqv, ck, cv, amask, pos_w, pos_b,
                                      kret, vret, attn);
  // 5) output conv (K=576, N=64)
  conv3x3_wmma<<<NB, 256, 91136, stream>>>(attn, attn, 64, 0,
                                           wout, out_b, outpre, 64, 0);
  // 6) residual + LayerNorm + LSTM update
  finalize_kernel<<<NB, 256, 0, stream>>>(outpre, input, gates, c_cur,
                                          ln_w, ln_b, h_next, c_next);
}
